// SimCLR_88673894793345
// MI455X (gfx1250) — matrix operand order repaired and checked by reading, not compiled, not run on hardware
//
#include <hip/hip_runtime.h>
#include <hip/hip_bf16.h>

typedef __attribute__((ext_vector_type(16))) _Float16 v16h;
typedef __attribute__((ext_vector_type(8)))  _Float16 v8h;
typedef __attribute__((ext_vector_type(8)))  float    v8f;

#define NROWS     8192
#define FEATD     256
#define NTILES    (NROWS / 16)      // 512 column tiles
#define KSTEPS    (FEATD / 32)      // 8 wmma K-steps of 32
#define SLICES    8                 // column split per row strip
#define TPS       (NTILES / SLICES) // 64 tiles per slice
#define RT_PER_BLK 8                // row tiles (waves) per block
#define LDS_STRIDE 264              // halves per LDS row (512B + 16B pad)
#define TINV      (1.0f / 0.07f)
#define ESCALE    (TINV * 1.44269504088896340736f)   // 1/T * log2(e)

// ---------------------------------------------------------------------------
// Kernel 1: row-normalize fp32 features -> f16 (one wave32 per row)
// ---------------------------------------------------------------------------
__global__ void simclr_normalize_f16(const float* __restrict__ feat,
                                     _Float16* __restrict__ out) {
    const int lane = threadIdx.x & 31;
    const int row  = (blockIdx.x * blockDim.x + threadIdx.x) >> 5;
    const float* r = feat + (size_t)row * FEATD;

    float v[8];
    float ss = 0.0f;
    #pragma unroll
    for (int i = 0; i < 8; ++i) {
        v[i] = r[lane + 32 * i];
        ss += v[i] * v[i];
    }
    #pragma unroll
    for (int m = 16; m >= 1; m >>= 1) ss += __shfl_xor(ss, m, 32);

    float inv = 1.0f / fmaxf(sqrtf(ss), 1e-12f);

    _Float16* o = out + (size_t)row * FEATD;
    #pragma unroll
    for (int i = 0; i < 8; ++i) o[lane + 32 * i] = (_Float16)(v[i] * inv);
}

// ---------------------------------------------------------------------------
// A-strip load (WMMA A layout), kept in registers for the whole kernel
// ---------------------------------------------------------------------------
__device__ __forceinline__ void load_a_strip(const _Float16* __restrict__ fh,
                                             int rbase, int l16, int half,
                                             v16h (&A)[KSTEPS]) {
    const _Float16* abase = fh + (size_t)(rbase + l16) * FEATD + half * 16;
    #pragma unroll
    for (int kk = 0; kk < KSTEPS; ++kk) {
        v8h lo = *(const v8h*)(abase + kk * 32);
        v8h hi = *(const v8h*)(abase + kk * 32 + 8);
        A[kk] = __builtin_shufflevector(lo, hi, 0,1,2,3,4,5,6,7,
                                                 8,9,10,11,12,13,14,15);
    }
}

// Direct-from-global B tile (diag correction only)
__device__ __forceinline__ void load_b_tile_global(const _Float16* __restrict__ fh,
                                                   int ct, int l16, int half,
                                                   v16h (&B)[KSTEPS]) {
    const _Float16* bbase = fh + (size_t)(ct * 16 + l16) * FEATD + half * 16;
    #pragma unroll
    for (int kk = 0; kk < KSTEPS; ++kk) {
        v8h lo = *(const v8h*)(bbase + kk * 32);
        v8h hi = *(const v8h*)(bbase + kk * 32 + 8);
        B[kk] = __builtin_shufflevector(lo, hi, 0,1,2,3,4,5,6,7,
                                                 8,9,10,11,12,13,14,15);
    }
}

__device__ __forceinline__ v8f gemm_tile(const v16h (&A)[KSTEPS],
                                         const v16h (&B)[KSTEPS]) {
    v8f c = {};
    #pragma unroll
    for (int kk = 0; kk < KSTEPS; ++kk)
        c = __builtin_amdgcn_wmma_f32_16x16x32_f16(
                false, A[kk], false, B[kk], (short)0, c, false, false);
    return c;
}

// ---------------------------------------------------------------------------
// Kernel 2: block = 8 row-tiles (one per wave) x one 64-tile column slice.
// B tiles staged in double-buffered LDS, shared by all 8 waves.
// ---------------------------------------------------------------------------
__global__ void __launch_bounds__(256)
simclr_gram_partial(const _Float16* __restrict__ fh,
                    float* __restrict__ partial,   // [NROWS][SLICES]
                    float* __restrict__ pos) {     // [NROWS]
    __shared__ _Float16 lds[2][16][LDS_STRIDE];   // 2 x 8.25 KB

    const int tid   = threadIdx.x;
    const int lane  = tid & 31;
    const int wave  = tid >> 5;
    const int g     = blockIdx.x >> 3;            // row-tile group (0..63)
    const int slice = blockIdx.x & (SLICES - 1);
    const int rtile = g * RT_PER_BLK + wave;      // 0..511
    const int half  = lane >> 4;
    const int l16   = lane & 15;
    const int rbase = rtile * 16;
    const int ct0   = slice * TPS;

    // staging map: thread copies 32 contiguous bytes of one B-tile row
    const int srow = tid >> 4;                    // 0..15
    const int soff = (tid & 15) * 16;             // halves, 0..240 step 16

    v16h A[KSTEPS];
    load_a_strip(fh, rbase, l16, half, A);

    float sumacc[8];
    #pragma unroll
    for (int r = 0; r < 8; ++r) sumacc[r] = 0.0f;

    // ---- stage / compute lambdas -----------------------------------------
    auto stage = [&](int buf, int ct) {
        const _Float16* src = fh + (size_t)(ct * 16 + srow) * FEATD + soff;
        v8h a0 = *(const v8h*)(src);
        v8h a1 = *(const v8h*)(src + 8);
        *(v8h*)&lds[buf][srow][soff]     = a0;
        *(v8h*)&lds[buf][srow][soff + 8] = a1;
    };
    auto compute = [&](int buf) {
        v8f c = {};
        #pragma unroll
        for (int kk = 0; kk < KSTEPS; ++kk) {
            const _Float16* bp = &lds[buf][l16][kk * 32 + half * 16];
            v8h lo = *(const v8h*)(bp);
            v8h hi = *(const v8h*)(bp + 8);
            v16h B = __builtin_shufflevector(lo, hi, 0,1,2,3,4,5,6,7,
                                                      8,9,10,11,12,13,14,15);
            c = __builtin_amdgcn_wmma_f32_16x16x32_f16(
                    false, A[kk], false, B, (short)0, c, false, false);
        }
        #pragma unroll
        for (int r = 0; r < 8; ++r)
            sumacc[r] += __builtin_amdgcn_exp2f(c[r] * ESCALE);
    };

    // ---- main loop: static double buffers, one barrier per tile ----------
    stage(0, ct0);
    __syncthreads();
    for (int t = 0; t < TPS; t += 2) {
        stage(1, ct0 + t + 1);          // TPS even -> t+1 always valid
        compute(0);                     // tile t
        __syncthreads();

        const int n2 = (t + 2 < TPS) ? (t + 2) : (t + 1); // last iter: redundant
        stage(0, ct0 + n2);
        compute(1);                     // tile t+1
        __syncthreads();
    }

    // ---- diagonal correction (wave-uniform condition) --------------------
    float posacc[8];
    #pragma unroll
    for (int r = 0; r < 8; ++r) posacc[r] = 0.0f;

    const bool owns_diag = (slice == (rtile / TPS));
    if (owns_diag) {
        v16h Bd[KSTEPS];
        load_b_tile_global(fh, rtile, l16, half, Bd);
        v8f c = gemm_tile(A, Bd);      // bit-identical to in-loop result
        #pragma unroll
        for (int r = 0; r < 8; ++r) {
            const int M = r + half * 8;
            const float s = c[r];
            if (l16 == M)       sumacc[r] -= __builtin_amdgcn_exp2f(s * ESCALE);
            if (l16 == (M ^ 1)) posacc[r] += s;      // exactly one lane
        }
    }

    // ---- reduce each row over its 16-lane group --------------------------
    #pragma unroll
    for (int m = 1; m < 16; m <<= 1) {
        #pragma unroll
        for (int r = 0; r < 8; ++r) {
            sumacc[r] += __shfl_xor(sumacc[r], m, 32);
            posacc[r] += __shfl_xor(posacc[r], m, 32);
        }
    }

    if (l16 == 0) {   // lanes 0 / 16 own rows M = r / r+8
        #pragma unroll
        for (int r = 0; r < 8; ++r) {
            const int row = rbase + r + half * 8;
            partial[(size_t)row * SLICES + slice] = sumacc[r];
            if (owns_diag) pos[row] = posacc[r];
        }
    }
}

// ---------------------------------------------------------------------------
// Kernel 3: combine partials, per-row loss, deterministic mean
// ---------------------------------------------------------------------------
__global__ void simclr_finalize(const float* __restrict__ partial,
                                const float* __restrict__ pos,
                                float* __restrict__ out) {
    __shared__ float sm[256];
    float s = 0.0f;
    for (int row = threadIdx.x; row < NROWS; row += 256) {
        float t = 0.0f;
        #pragma unroll
        for (int k = 0; k < SLICES; ++k) t += partial[(size_t)row * SLICES + k];
        s += __logf(t) - pos[row] * TINV;
    }
    sm[threadIdx.x] = s;
    __syncthreads();
    for (int stride = 128; stride > 0; stride >>= 1) {
        if ((int)threadIdx.x < stride) sm[threadIdx.x] += sm[threadIdx.x + stride];
        __syncthreads();
    }
    if (threadIdx.x == 0) out[0] = sm[0] * (1.0f / (float)NROWS);
}

// ---------------------------------------------------------------------------
extern "C" void kernel_launch(void* const* d_in, const int* in_sizes, int n_in,
                              void* d_out, int out_size, void* d_ws, size_t ws_size,
                              hipStream_t stream) {
    (void)in_sizes; (void)n_in; (void)out_size; (void)ws_size;

    const float* feat = (const float*)d_in[0];
    char* ws          = (char*)d_ws;
    _Float16* fh      = (_Float16*)ws;                                  // 4 MB
    float* partial    = (float*)(ws + (size_t)NROWS * FEATD * 2);       // 256 KB
    float* pos        = (float*)(ws + (size_t)NROWS * FEATD * 2
                                    + (size_t)NROWS * SLICES * 4);      // 32 KB
    float* out        = (float*)d_out;

    simclr_normalize_f16<<<NROWS / 8, 256, 0, stream>>>(feat, fh);

    // 64 row-tile groups x 8 slices = 512 blocks x 8 waves = 4096 waves
    simclr_gram_partial<<<(NTILES / RT_PER_BLK) * SLICES, 256, 0, stream>>>
        (fh, partial, pos);

    simclr_finalize<<<1, 256, 0, stream>>>(partial, pos, out);
}
